// EdgePredNet_3942779978325
// MI455X (gfx1250) — matrix operand order repaired and checked by reading, not compiled, not run on hardware
//
#include <hip/hip_runtime.h>

typedef __bf16 bf16_t;
typedef __attribute__((ext_vector_type(16))) __bf16 v16bf;
typedef __attribute__((ext_vector_type(8)))  __bf16 v8bf;
typedef __attribute__((ext_vector_type(4)))  __bf16 v4bf;
typedef __attribute__((ext_vector_type(8)))  float  v8f;

// ---- shared layout (bytes) for both workspace (global) and LDS ----
#define OFF_PW1 0        // W1 packed bf16 B-fragments: 8 kt * 8 nt * 32 lanes * 16 halves
#define OFF_PW2 65536    // W2 packed: 4 kt * 8 nt
#define OFF_PW3 98304    // W3 packed: 4 kt * 8 nt
#define OFF_B1  131072   // fp32 x128
#define OFF_B2  131584
#define OFF_B3  132096
#define OFF_W4  132608   // fp32 x128
#define OFF_B4  133120   // fp32 x1 (padded to 16B)
#define W_BYTES 133136
#define OFF_X   133136   // 256 rows x 128 cols bf16: x_i, then x_j, then layer-2 out
#define OFF_H   198672   // 256 rows x 128 cols bf16: layer-1 out
#define SMEM_TOTAL 264208

#define EDGES_PER_BLOCK 256   // 8 waves x 32 edges (2 m-tiles per wave)

// ---------------- kernel 1: weight convert + B-fragment pack ----------------
// B-fragment layout for v_wmma_*_16x16x32_bf16 (ISA 7.12.2):
//   element j of lane L = W[K = kt*32 + (L<16?0:8) + (j&7) + (j>=8?16:0)][N = nt*16 + (L&15)]
// Packed so each lane reads 16 contiguous halves (32B) per fragment.
__device__ inline void pack_tile(const float* __restrict__ W, bf16_t* __restrict__ dst,
                                 int idx, int ldn) {
  int f = idx >> 9, r = idx & 511;
  int lane = r >> 4, j = r & 15;
  int kt = f >> 3, nt = f & 7;
  int K = kt * 32 + ((lane >> 4) << 3) + (j & 7) + ((j >> 3) << 4);
  int N = nt * 16 + (lane & 15);
  dst[idx] = (bf16_t)W[K * ldn + N];
}

__global__ __launch_bounds__(256) void pack_weights(
    const float* __restrict__ W1, const float* __restrict__ b1,
    const float* __restrict__ W2, const float* __restrict__ b2,
    const float* __restrict__ W3, const float* __restrict__ b3,
    const float* __restrict__ W4, const float* __restrict__ b4,
    char* __restrict__ ws) {
  int tid = blockIdx.x * 256 + threadIdx.x;
  if (tid < 32768) {
    pack_tile(W1, (bf16_t*)(ws + OFF_PW1), tid, 128);
  } else if (tid < 49152) {
    pack_tile(W2, (bf16_t*)(ws + OFF_PW2), tid - 32768, 128);
  } else if (tid < 65536) {
    pack_tile(W3, (bf16_t*)(ws + OFF_PW3), tid - 49152, 128);
  } else {
    int r = tid - 65536;
    if (r < 128)       ((float*)(ws + OFF_B1))[r]       = b1[r];
    else if (r < 256)  ((float*)(ws + OFF_B2))[r - 128] = b2[r - 128];
    else if (r < 384)  ((float*)(ws + OFF_B3))[r - 256] = b3[r - 256];
    else if (r < 512)  ((float*)(ws + OFF_W4))[r - 384] = W4[r - 384];
    else if (r == 512) ((float*)(ws + OFF_B4))[0]       = b4[0];
  }
}

// ---------------- kernel 2: fused 4-layer edge MLP ----------------
__device__ inline v16bf cat16(v8bf lo, v8bf hi) {
  return __builtin_shufflevector(lo, hi, 0,1,2,3,4,5,6,7,8,9,10,11,12,13,14,15);
}

#define WMMA_BF16(a, b, c) \
  __builtin_amdgcn_wmma_f32_16x16x32_bf16(false, (a), false, (b), (short)0, (c), false, false)

// Dual-m-tile GEMM slab: rows [m0, m0+32) x packed-B -> acc0/acc1 (each 16x128 f32).
// Each B fragment (2x ds_load_b128) now feeds TWO WMMAs -> halves LDS B-traffic/FLOP.
// A-fragment layout (ISA 7.12.2, 16-bit A 16x32):
//   lane L, row = m + (L&15); a[0..7] = K kb..kb+7, a[8..15] = K kb+16..kb+23,
//   kb = kt*32 + (L<16 ? 0 : 8).
__device__ inline void wmma_slab(const bf16_t* __restrict__ src, int lds,
                                 const bf16_t* __restrict__ w, int KT, int wktoff,
                                 int lane, int m0, v8f* acc0, v8f* acc1) {
  const int arow = m0 + (lane & 15);
  const int koff = (lane >> 4) << 3;
  for (int kt = 0; kt < KT; ++kt) {
    const bf16_t* ap0 = src + arow * lds + kt * 32 + koff;
    const bf16_t* ap1 = ap0 + 16 * lds;
    v16bf a0 = cat16(*(const v8bf*)ap0, *(const v8bf*)(ap0 + 16));
    v16bf a1 = cat16(*(const v8bf*)ap1, *(const v8bf*)(ap1 + 16));
#pragma unroll
    for (int nt = 0; nt < 8; ++nt) {
      v16bf b = *(const v16bf*)(w + (((kt + wktoff) * 8 + nt) * 32 + lane) * 16);
      acc0[nt] = WMMA_BF16(a0, b, acc0[nt]);
      acc1[nt] = WMMA_BF16(a1, b, acc1[nt]);
    }
  }
}

// C/D layout (ISA 7.12.2, 32-bit C/D 16x16): VGPR r, lanes 0-15 -> row r,
// lanes 16-31 -> row r+8; col = nt*16 + (lane&15).
__device__ inline void relu_store(const v8f* acc, const float* __restrict__ bias,
                                  bf16_t* __restrict__ dst, int ldd,
                                  int lane, int m0) {
  const int col = lane & 15;
  const int rbase = m0 + ((lane >> 4) << 3);
#pragma unroll
  for (int nt = 0; nt < 8; ++nt) {
    float bv = bias[nt * 16 + col];
#pragma unroll
    for (int r = 0; r < 8; ++r) {
      float v = acc[nt][r] + bv;
      v = v > 0.f ? v : 0.f;
      dst[(rbase + r) * ldd + nt * 16 + col] = (bf16_t)v;
    }
  }
}

// Gather 32 node-feature rows (one half of the edge pair) -> sX rows [m0, m0+32).
// Per row: 32 lanes x float4 = one coalesced 512B read (x is 51MB, L2-resident).
__device__ inline void gather_half(const float* __restrict__ x,
                                   const long long* __restrict__ eidx,
                                   bf16_t* __restrict__ sX, int lane, int m0,
                                   long long ebase, int E, int half) {
  for (int r = 0; r < 32; ++r) {
    const int row = m0 + r;
    long long ge = ebase + row;
    if (ge >= E) ge = (long long)E - 1;     // clamp for partial last block
    const long long idx = eidx[(long long)half * E + ge];
    float4 a = *(const float4*)(x + idx * 128 + lane * 4);
    v4bf pa = {(bf16_t)a.x, (bf16_t)a.y, (bf16_t)a.z, (bf16_t)a.w};
    *(v4bf*)(sX + row * 128 + lane * 4) = pa;
  }
}

__global__ __launch_bounds__(256, 1) void edge_mlp(
    const float* __restrict__ x, const long long* __restrict__ eidx,
    const char* __restrict__ ws, float* __restrict__ out, int E) {
  extern __shared__ __align__(16) char smem[];
  bf16_t* sX = (bf16_t*)(smem + OFF_X);
  bf16_t* sH = (bf16_t*)(smem + OFF_H);

  const int tid  = threadIdx.x;
  const int lane = tid & 31;
  const int wave = tid >> 5;

  // Stage packed weights + biases global -> LDS (130 KB, 16B chunks).
  {
    const uint4* g = (const uint4*)ws;
    uint4* l = (uint4*)smem;
    for (int i = tid; i < (W_BYTES >> 4); i += 256) l[i] = g[i];
  }
  __syncthreads();  // only barrier: afterwards each wave touches only its own rows

  const int m0 = wave * 32;            // this wave's 32 edges (2 m-tiles)
  const long long ebase = (long long)blockIdx.x * EDGES_PER_BLOCK;

  const v8f vzero = {0.f,0.f,0.f,0.f,0.f,0.f,0.f,0.f};
  v8f acc0[8], acc1[8];
#pragma unroll
  for (int nt = 0; nt < 8; ++nt) { acc0[nt] = vzero; acc1[nt] = vzero; }

  // ---- Layer 1 in two K-phases so sX (64KB) is reused for x_i then x_j ----
  gather_half(x, eidx, sX, lane, m0, ebase, E, 0);                       // x_i
  wmma_slab(sX, 128, (const bf16_t*)(smem + OFF_PW1), 4, 0, lane, m0, acc0, acc1);
  gather_half(x, eidx, sX, lane, m0, ebase, E, 1);                       // x_j
  wmma_slab(sX, 128, (const bf16_t*)(smem + OFF_PW1), 4, 4, lane, m0, acc0, acc1);
  relu_store(acc0, (const float*)(smem + OFF_B1), sH, 128, lane, m0);
  relu_store(acc1, (const float*)(smem + OFF_B1), sH, 128, lane, m0 + 16);

  // ---- Layer 2: sH -> sX (wave-private rows, no barrier needed) ----
#pragma unroll
  for (int nt = 0; nt < 8; ++nt) { acc0[nt] = vzero; acc1[nt] = vzero; }
  wmma_slab(sH, 128, (const bf16_t*)(smem + OFF_PW2), 4, 0, lane, m0, acc0, acc1);
  relu_store(acc0, (const float*)(smem + OFF_B2), sX, 128, lane, m0);
  relu_store(acc1, (const float*)(smem + OFF_B2), sX, 128, lane, m0 + 16);

  // ---- Layer 3: stays in registers ----
#pragma unroll
  for (int nt = 0; nt < 8; ++nt) { acc0[nt] = vzero; acc1[nt] = vzero; }
  wmma_slab(sX, 128, (const bf16_t*)(smem + OFF_PW3), 4, 0, lane, m0, acc0, acc1);

  // ---- Layer 4 fused: bias+relu in regs, dot with W4, 16-lane reduce, sigmoid ----
  {
    const int col = lane & 15;
    const float* b3v = (const float*)(smem + OFF_B3);
    const float* w4  = (const float*)(smem + OFF_W4);
    const float  b4v = ((const float*)(smem + OFF_B4))[0];
    float wv[8], bb[8];
#pragma unroll
    for (int nt = 0; nt < 8; ++nt) { wv[nt] = w4[nt * 16 + col]; bb[nt] = b3v[nt * 16 + col]; }
#pragma unroll
    for (int t = 0; t < 2; ++t) {
      const v8f* acc = t ? acc1 : acc0;
      const int mb = m0 + t * 16;
#pragma unroll
      for (int r = 0; r < 8; ++r) {
        float p = 0.f;
#pragma unroll
        for (int nt = 0; nt < 8; ++nt) {
          float h = acc[nt][r] + bb[nt];
          h = h > 0.f ? h : 0.f;
          p += h * wv[nt];
        }
        // rows r (lanes 0-15) and r+8 (lanes 16-31): reduce within each 16-lane half
        p += __shfl_xor(p, 1);
        p += __shfl_xor(p, 2);
        p += __shfl_xor(p, 4);
        p += __shfl_xor(p, 8);
        const int row = mb + ((lane >> 4) << 3) + r;
        const long long ge = ebase + row;
        if (col == 0 && ge < (long long)E) {
          out[ge] = 1.f / (1.f + __expf(-(p + b4v)));
        }
      }
    }
  }
}

// ---------------- launch ----------------
extern "C" void kernel_launch(void* const* d_in, const int* in_sizes, int n_in,
                              void* d_out, int out_size, void* d_ws, size_t ws_size,
                              hipStream_t stream) {
  const float*     x    = (const float*)d_in[0];
  const long long* eidx = (const long long*)d_in[1];   // edge_index is int64
  const float* W1 = (const float*)d_in[2];
  const float* b1 = (const float*)d_in[3];
  const float* W2 = (const float*)d_in[4];
  const float* b2 = (const float*)d_in[5];
  const float* W3 = (const float*)d_in[6];
  const float* b3 = (const float*)d_in[7];
  const float* W4 = (const float*)d_in[8];
  const float* b4 = (const float*)d_in[9];
  char*  ws  = (char*)d_ws;
  float* out = (float*)d_out;
  const int E = in_sizes[1] / 2;   // (2, E) int64

  hipFuncSetAttribute(reinterpret_cast<const void*>(edge_mlp),
                      hipFuncAttributeMaxDynamicSharedMemorySize, SMEM_TOTAL);

  pack_weights<<<259, 256, 0, stream>>>(W1, b1, W2, b2, W3, b3, W4, b4, ws);

  const int blocks = (E + EDGES_PER_BLOCK - 1) / EDGES_PER_BLOCK;
  edge_mlp<<<blocks, 256, SMEM_TOTAL, stream>>>(x, eidx, ws, out, E);
}
